// Classifier_1821066133734
// MI455X (gfx1250) — matrix operand order repaired and checked by reading, not compiled, not run on hardware
//
#include <hip/hip_runtime.h>
#include <hip/hip_bf16.h>

// ---------------------------------------------------------------------------
// GDAS cell forward for MI455X (gfx1250, wave32, WMMA).
//   B=16384, C_IN=1024, NODE_NUM=6, 21 edges, 9 ops, TAU=10, EPS=1e-5.
// Per-edge op chosen on-device (argmax of gumbel-softmax logits); each branch
// kernel reads op_index[edge] and exits uniformly if not selected. hardwts at
// the argmax is numerically exactly 1.0 (one_hot - prob + prob), so no scale.
// GEMMs: bf16 WMMA (v_wmma_f32_16x16x32_bf16), fp32 accumulate, software-
// pipelined global->LDS staging with double buffering + global prefetch.
// Block tile 128x128 (8 waves, each 32x64 = 2x4 WMMA subtiles), K-step 32.
// Pool ops: analytic fold of BN(BN(.)/9) -> single reduction + affine apply.
// ---------------------------------------------------------------------------

typedef __bf16 bf16_t;
typedef bf16_t v16bf __attribute__((ext_vector_type(16)));
typedef bf16_t v2bf  __attribute__((ext_vector_type(2)));
typedef float  v8f   __attribute__((ext_vector_type(8)));

#define BN_EPS 1e-5f
#define BATCH  16384
#define COUT   512

union FragU { v16bf v; uint4 q[2]; };

// native packed f32x2 -> bf16x2 (v_cvt_pk_bf16_f32)
__device__ __forceinline__ unsigned pk_bf16(float x, float y) {
    v2bf p;
    p.x = (bf16_t)x;
    p.y = (bf16_t)y;
    return __builtin_bit_cast(unsigned, p);
}

// ---------------------------------------------------------------------------
// op selection: logits = (log_softmax(arch) + gumbel)/TAU ; argmax per edge
// ---------------------------------------------------------------------------
__global__ void select_ops_kernel(const float* __restrict__ ap,
                                  const float* __restrict__ gum,
                                  int* __restrict__ opidx) {
    int e = threadIdx.x;
    if (e >= 21) return;
    float mx = -1e30f;
    for (int k = 0; k < 9; ++k) mx = fmaxf(mx, ap[e * 9 + k]);
    float s = 0.f;
    for (int k = 0; k < 9; ++k) s += expf(ap[e * 9 + k] - mx);
    float lse = mx + logf(s);
    int best = 0; float bv = -1e30f;
    for (int k = 0; k < 9; ++k) {
        float logit = (ap[e * 9 + k] - lse + gum[e * 9 + k]) * 0.1f; // /TAU
        if (logit > bv) { bv = logit; best = k; }
    }
    opidx[e] = best;
}

__global__ void fill_zero_kernel(float* __restrict__ p, long long n) {
    long long i = (long long)blockIdx.x * blockDim.x + threadIdx.x;
    long long stride = (long long)gridDim.x * blockDim.x;
    for (; i < n; i += stride) p[i] = 0.f;
}

// ---------------------------------------------------------------------------
// Generic edge GEMM:  Out[:, outCol+n] (+)= act( A[:, aCol+k] @ W[n,k]^T + b )
// kind 0: dense (ops 2..4, act = op-2)
// kind 1: grouped dense (ops 5..7), blockIdx.z = group
// kind 2: pool preprocess (ops 0..1): z = relu(A) @ pw[p]^T, store (no acc)
// Block: 256 thr (8 waves, 4M x 2N), tile 128(M) x 128(N), K-step 32,
// double-buffered LDS staging, per-wave 2x4 WMMA subtiles (32x64).
// ---------------------------------------------------------------------------
__global__ __launch_bounds__(256)
void gemm_edge_kernel(const int* __restrict__ opidx, int edge, int kind,
                      const float* __restrict__ A, int lda,
                      const float* __restrict__ W, long long wPerA, long long wPerG,
                      const float* __restrict__ bias, int biasPerA, int biasPerG,
                      float* __restrict__ Out, int ldo, int outColPerG, int aColPerG,
                      int K) {
    const int op   = opidx[edge];
    const int opLo = (kind == 0) ? 2 : (kind == 1 ? 5 : 0);
    const int opHi = (kind == 0) ? 4 : (kind == 1 ? 7 : 1);
    if (op < opLo || op > opHi) return;          // uniform: whole grid exits
    const int  a        = op - opLo;             // act idx (dense/group) or p (pool)
    const int  act      = (kind == 2) ? 3 : a;   // 3 = identity
    const bool prerelu  = (kind == 2);
    const bool storeOut = (kind == 2);

    const int g = blockIdx.z;
    const float* Wg = W + (long long)a * wPerA + (long long)g * wPerG;
    const float* Bg = bias ? (bias + a * biasPerA + g * biasPerG) : nullptr;
    const int aCol    = g * aColPerG;
    const int outColG = g * outColPerG;
    const int rowBase = blockIdx.x * 128;
    const int nBase   = blockIdx.y * 128;

    __shared__ __align__(16) unsigned short sA[2][128 * 32];
    __shared__ __align__(16) unsigned short sB[2][128 * 32];

    const int tid  = threadIdx.x;
    const int lane = tid & 31;
    const int wave = tid >> 5;
    const int wm   = wave & 3;      // 4 waves along M (32 rows each)
    const int wn   = wave >> 2;     // 2 waves along N (64 cols each)
    const int half = lane >> 4;
    const int l16  = lane & 15;

    v8f acc[2][4];
    #pragma unroll
    for (int i = 0; i < 2; ++i)
        #pragma unroll
        for (int j = 0; j < 4; ++j)
            #pragma unroll
            for (int v = 0; v < 8; ++v) acc[i][j][v] = 0.f;

    float4 ar[4], wr[4];   // in-flight staging registers

    auto issue_loads = [&](int k0) {
        #pragma unroll
        for (int it = 0; it < 4; ++it) {
            int idx = tid + it * 256;            // 1024 float4 slots (128x32)
            int r   = idx >> 3;
            int c4  = (idx & 7) << 2;
            ar[it] = *(const float4*)&A[(size_t)(rowBase + r) * lda + aCol + k0 + c4];
            wr[it] = *(const float4*)&Wg[(size_t)(nBase + r) * K + k0 + c4];
        }
    };
    auto store_tiles = [&](int buf) {
        #pragma unroll
        for (int it = 0; it < 4; ++it) {
            int idx = tid + it * 256;
            int r   = idx >> 3;
            int c4  = (idx & 7) << 2;
            float4 f = ar[it];
            if (prerelu) {
                f.x = fmaxf(f.x, 0.f); f.y = fmaxf(f.y, 0.f);
                f.z = fmaxf(f.z, 0.f); f.w = fmaxf(f.w, 0.f);
            }
            uint2 p;
            p.x = pk_bf16(f.x, f.y);
            p.y = pk_bf16(f.z, f.w);
            *(uint2*)&sA[buf][r * 32 + c4] = p;
            uint2 q;
            q.x = pk_bf16(wr[it].x, wr[it].y);
            q.y = pk_bf16(wr[it].z, wr[it].w);
            *(uint2*)&sB[buf][r * 32 + c4] = q;
        }
    };

    // prologue: stage K-tile 0 into buffer 0
    issue_loads(0);
    store_tiles(0);
    __syncthreads();

    const int nIter = K >> 5;
    for (int iter = 0; iter < nIter; ++iter) {
        const int  cur  = iter & 1;
        const bool more = (iter + 1 < nIter);

        // issue next tile's global loads early (latency hidden under WMMAs)
        if (more) issue_loads((iter + 1) << 5);

        // prefetch two tiles ahead: one prefetch per 128B row-segment
        if (iter + 2 < nIter && (tid & 7) == 0) {
            int k2 = (iter + 2) << 5;
            #pragma unroll
            for (int it = 0; it < 4; ++it) {
                int r = (tid + it * 256) >> 3;
                __builtin_prefetch(&A[(size_t)(rowBase + r) * lda + aCol + k2], 0, 1);
                __builtin_prefetch(&Wg[(size_t)(nBase + r) * K + k2], 0, 1);
            }
        }

        // fragments: contiguous k order (k = half*16 + s) for both A and B (wave32)
        FragU af[2], bfr[4];
        #pragma unroll
        for (int tm = 0; tm < 2; ++tm) {
            int row = wm * 32 + tm * 16 + l16;
            const uint4* pr = (const uint4*)&sA[cur][row * 32];
            af[tm].q[0] = pr[half * 2];     // k = half*16 + 0..7
            af[tm].q[1] = pr[half * 2 + 1]; // k = half*16 + 8..15
        }
        #pragma unroll
        for (int tn = 0; tn < 4; ++tn) {
            int col = wn * 64 + tn * 16 + l16;
            const uint4* pc = (const uint4*)&sB[cur][col * 32];
            bfr[tn].q[0] = pc[half * 2];        // k = half*16 + 0..7
            bfr[tn].q[1] = pc[half * 2 + 1];    // k = half*16 + 8..15
        }
        #pragma unroll
        for (int tm = 0; tm < 2; ++tm)
            #pragma unroll
            for (int tn = 0; tn < 4; ++tn)
                acc[tm][tn] = __builtin_amdgcn_wmma_f32_16x16x32_bf16(
                    false, af[tm].v, false, bfr[tn].v,
                    (short)0, acc[tm][tn], false, false);

        // convert + stage next tile into the other buffer, then one barrier.
        // (Writes target buf cur^1; all reads of that buffer happened before
        //  the barrier of the previous iteration, so no WAR hazard.)
        if (more) store_tiles(cur ^ 1);
        __syncthreads();
    }

    // --- epilogue: bias + act, accumulate (or store for pool-pre) ---
    #pragma unroll
    for (int tm = 0; tm < 2; ++tm) {
        #pragma unroll
        for (int tn = 0; tn < 4; ++tn) {
            int nLocal = nBase + wn * 64 + tn * 16 + l16;   // weight/bias row
            int outN   = outColG + nLocal;
            float bv = Bg ? Bg[nLocal] : 0.f;
            #pragma unroll
            for (int v = 0; v < 8; ++v) {
                int m = rowBase + wm * 32 + tm * 16 + half * 8 + v;
                float x = acc[tm][tn][v] + bv;
                if (act == 0)      x = fmaxf(x, 0.f);
                else if (act == 1) x = 1.f / (1.f + expf(-x));
                else if (act == 2) x = tanhf(x);
                size_t o = (size_t)m * ldo + outN;
                if (storeOut) Out[o] = x; else Out[o] += x;
            }
        }
    }
}

// ---------------------------------------------------------------------------
// Pool batch stats: per-channel sum / sumsq over 16384 rows x 512 channels.
// Thread t owns channels {2t, 2t+1}; block handles 64 rows; atomics combine.
// ---------------------------------------------------------------------------
__global__ __launch_bounds__(256)
void pool_reduce_kernel(const int* __restrict__ opidx, int edge,
                        const float* __restrict__ X, int ldx,
                        float* __restrict__ sum, float* __restrict__ sumsq) {
    if (opidx[edge] > 1) return;
    int t = threadIdx.x;
    int c = t * 2;
    long long r0 = (long long)blockIdx.x * 64;
    float s0 = 0.f, s1 = 0.f, q0 = 0.f, q1 = 0.f;
    for (int r = 0; r < 64; ++r) {
        float2 v = *(const float2*)&X[(r0 + r) * ldx + c];
        s0 += v.x; s1 += v.y;
        q0 += v.x * v.x; q1 += v.y * v.y;
    }
    atomicAdd(&sum[c], s0);   atomicAdd(&sum[c + 1], s1);
    atomicAdd(&sumsq[c], q0); atomicAdd(&sumsq[c + 1], q1);
}

// ---------------------------------------------------------------------------
// Pool finalize: per-channel (mean, scale) from (sum, sumsq).
//   A-edge: S = a*rs / sqrt(a^2*v*rs^2 + eps), rs = rsqrt(v+eps)
//   B-edge: S = a / sqrt(a^2*v + eps)
// a = 1/9 for avg-pool (op 0), 1 for max-pool (op 1). Overwrites in place:
//   sum[c] <- mean, sumsq[c] <- scale.
// ---------------------------------------------------------------------------
__global__ void pool_finalize_kernel(const int* __restrict__ opidx, int edge,
                                     float* __restrict__ sum,
                                     float* __restrict__ sumsq, int isA) {
    int op = opidx[edge];
    if (op > 1) return;
    int c = (int)(blockIdx.x * blockDim.x + threadIdx.x);
    if (c >= COUT) return;
    float aSc  = (op == 0) ? (1.f / 9.f) : 1.f;
    float m = sum[c] * (1.f / (float)BATCH);
    float v = fmaxf(sumsq[c] * (1.f / (float)BATCH) - m * m, 0.f);
    float S;
    if (isA) {
        float rs = rsqrtf(v + BN_EPS);
        float t  = aSc * rs;
        S = t * rsqrtf(t * t * v + BN_EPS);
    } else {
        S = aSc * rsqrtf(aSc * aSc * v + BN_EPS);
    }
    sum[c]   = m;
    sumsq[c] = S;
}

// Pool apply: Out += (X - mean[c]) * scale[c]
__global__ __launch_bounds__(256)
void pool_apply_kernel(const int* __restrict__ opidx, int edge,
                       const float* __restrict__ X,
                       const float* __restrict__ mean, const float* __restrict__ scale,
                       float* __restrict__ Out) {
    if (opidx[edge] > 1) return;
    long long n = (long long)BATCH * COUT;
    long long i = (long long)blockIdx.x * blockDim.x + threadIdx.x;
    long long stride = (long long)gridDim.x * blockDim.x;
    for (; i < n; i += stride) {
        int c = (int)(i & (COUT - 1));
        Out[i] += (X[i] - mean[c]) * scale[c];
    }
}

// ---------------------------------------------------------------------------
extern "C" void kernel_launch(void* const* d_in, const int* in_sizes, int n_in,
                              void* d_out, int out_size, void* d_ws, size_t ws_size,
                              hipStream_t stream) {
    const float* x   = (const float*)d_in[0];
    const float* ap  = (const float*)d_in[1];
    const float* gum = (const float*)d_in[2];
    const float* dwA = (const float*)d_in[3];
    const float* dbA = (const float*)d_in[4];
    const float* gwA = (const float*)d_in[5];
    const float* gbA = (const float*)d_in[6];
    const float* pwA = (const float*)d_in[7];
    const float* dwB = (const float*)d_in[8];
    const float* dbB = (const float*)d_in[9];
    const float* gwB = (const float*)d_in[10];
    const float* gbB = (const float*)d_in[11];

    float* ws      = (float*)d_ws;
    int*   opidx   = (int*)ws;                 // 64 slots reserved
    float* red_sum = ws + 64;                  // 512 (sum -> mean)
    float* red_sq  = ws + 64 + 512;            // 512 (sumsq -> scale)
    float* base    = ws + 64 + 1024;
    const size_t SN = (size_t)BATCH * COUT;    // 8.39M floats per state
    float* st[7];
    st[0] = nullptr;                           // state0 is x (cin=1024)
    for (int k = 1; k <= 5; ++k) st[k] = base + (size_t)(k - 1) * SN;
    st[6] = (float*)d_out;
    float* ztmp = base + 5 * SN;               // pool-pre temp [B,512]

    select_ops_kernel<<<1, 32, 0, stream>>>(ap, gum, opidx);

    int a_ct = 0, b_ct = 0;
    for (int i = 1; i <= 6; ++i) {
        fill_zero_kernel<<<2048, 256, 0, stream>>>(st[i], (long long)SN);
        for (int j = 0; j < i; ++j) {
            int e = j * 6 - j * (j - 1) / 2 + (i - j - 1);   // edge2index('j->i')
            if (j == 0) {
                // ---- A edges: cin=1024, cout=512 ----
                // dense (ops 2..4)
                gemm_edge_kernel<<<dim3(128, 4, 1), 256, 0, stream>>>(
                    opidx, e, 0, x, 1024,
                    dwA + (size_t)a_ct * 3 * 512 * 1024, 512LL * 1024, 0LL,
                    dbA + (size_t)a_ct * 3 * 512, 512, 0,
                    st[i], 512, 0, 0, 1024);
                // grouped dense (ops 5..7): 4 groups, Kg=256, coutg=128
                gemm_edge_kernel<<<dim3(128, 1, 4), 256, 0, stream>>>(
                    opidx, e, 1, x, 1024,
                    gwA + (size_t)a_ct * 3 * 4 * 128 * 256, 4LL * 128 * 256, 128LL * 256,
                    gbA + (size_t)a_ct * 3 * 512, 512, 128,
                    st[i], 512, 128, 256, 256);
                // pool preprocess (ops 0..1): z = relu(x) @ pw[p]^T
                gemm_edge_kernel<<<dim3(128, 4, 1), 256, 0, stream>>>(
                    opidx, e, 2, x, 1024,
                    pwA + (size_t)a_ct * 2 * 512 * 1024, 512LL * 1024, 0LL,
                    nullptr, 0, 0,
                    ztmp, 512, 0, 0, 1024);
                fill_zero_kernel<<<4, 256, 0, stream>>>(red_sum, 1024);
                pool_reduce_kernel<<<256, 256, 0, stream>>>(opidx, e, ztmp, 512,
                                                            red_sum, red_sq);
                pool_finalize_kernel<<<2, 256, 0, stream>>>(opidx, e, red_sum, red_sq, 1);
                pool_apply_kernel<<<4096, 256, 0, stream>>>(opidx, e, ztmp,
                                                            red_sum, red_sq, st[i]);
                ++a_ct;
            } else {
                // ---- B edges: cin=cout=512 ----
                const float* Ain = st[j];
                gemm_edge_kernel<<<dim3(128, 4, 1), 256, 0, stream>>>(
                    opidx, e, 0, Ain, 512,
                    dwB + (size_t)b_ct * 3 * 512 * 512, 512LL * 512, 0LL,
                    dbB + (size_t)b_ct * 3 * 512, 512, 0,
                    st[i], 512, 0, 0, 512);
                gemm_edge_kernel<<<dim3(128, 1, 4), 256, 0, stream>>>(
                    opidx, e, 1, Ain, 512,
                    gwB + (size_t)b_ct * 3 * 4 * 128 * 128, 4LL * 128 * 128, 128LL * 128,
                    gbB + (size_t)b_ct * 3 * 512, 512, 128,
                    st[i], 512, 128, 128, 128);
                // pool (ops 0..1) on B edges: BN of (a * state)
                fill_zero_kernel<<<4, 256, 0, stream>>>(red_sum, 1024);
                pool_reduce_kernel<<<256, 256, 0, stream>>>(opidx, e, Ain, 512,
                                                            red_sum, red_sq);
                pool_finalize_kernel<<<2, 256, 0, stream>>>(opidx, e, red_sum, red_sq, 0);
                pool_apply_kernel<<<4096, 256, 0, stream>>>(opidx, e, Ain,
                                                            red_sum, red_sq, st[i]);
                ++b_ct;
            }
        }
    }
}
